// RecurrentClassifier_13048110645603
// MI455X (gfx1250) — matrix operand order repaired and checked by reading, not compiled, not run on hardware
//
#include <hip/hip_runtime.h>
#include <hip/hip_bf16.h>

#define DEVINL __device__ __forceinline__

typedef __attribute__((ext_vector_type(16))) __bf16 v16bf;
typedef __attribute__((ext_vector_type(8)))  __bf16 v8bf;
typedef __attribute__((ext_vector_type(8)))  float  v8f;
typedef __attribute__((ext_vector_type(4)))  int    v4i;

constexpr int Bc  = 256;   // batch
constexpr int Dc  = 512;   // input dim
constexpr int Hc  = 1024;  // hidden
constexpr int NCc = 1000;  // classes
constexpr int Tc  = 64;    // timesteps
constexpr int Kc  = 6;     // ticks
constexpr int G4  = 4096;  // 4*Hc

// ---------------- workspace layout (bytes) ----------------
constexpr size_t OFF_WIHT  = 0;                               // [Dc][G4] bf16
constexpr size_t OFF_WHHT  = OFF_WIHT  + (size_t)Dc * G4 * 2; // [Hc][G4] bf16
constexpr size_t OFF_WDECT = OFF_WHHT  + (size_t)Hc * G4 * 2; // [Hc][1024] bf16
constexpr size_t OFF_XBF   = OFF_WDECT + (size_t)Hc * 1024 * 2; // [Tc][Bc][Dc] bf16
constexpr size_t OFF_BSUM  = OFF_XBF   + (size_t)Tc * Bc * Dc * 2; // [G4] f32
constexpr size_t OFF_BDEC  = OFF_BSUM  + (size_t)G4 * 4;       // [1024] f32
constexpr size_t OFF_XPROJ = OFF_BDEC  + 1024 * 4;             // [Bc][G4] f32
constexpr size_t OFF_H0    = OFF_XPROJ + (size_t)Bc * G4 * 4;  // [Bc][Hc] bf16
constexpr size_t OFF_H1    = OFF_H0    + (size_t)Bc * Hc * 2;  // [Bc][Hc] bf16
constexpr size_t OFF_CCUR  = OFF_H1    + (size_t)Bc * Hc * 2;  // [Bc][Hc] f32
constexpr size_t OFF_SH    = OFF_CCUR  + (size_t)Bc * Hc * 4;  // [Kc][Bc][Hc] f32
constexpr size_t OFF_CH    = OFF_SH    + (size_t)Kc * Bc * Hc * 4;
constexpr size_t OFF_YH    = OFF_CH    + (size_t)Kc * Bc * Hc * 4; // [Kc][Bc][1024] f32
constexpr size_t OFF_PH    = OFF_YH    + (size_t)Kc * Bc * 1024 * 4; // [Kc][Bc] f32

// output layout (flat floats): Y | P | N | Hs
constexpr size_t OUT_Y  = 0;
constexpr size_t OUT_P  = OUT_Y + (size_t)Bc * NCc * Tc;
constexpr size_t OUT_N  = OUT_P + Bc;
constexpr size_t OUT_HS = OUT_N + (size_t)Bc * Tc;

DEVINL float sigf(float x) { return 1.0f / (1.0f + __expf(-x)); }

// A fragment (16x32 bf16, M x K): lane<16 -> M=lane, K {k0..k0+7, k0+16..k0+23};
// lane>=16 -> M=lane-16, K {k0+8..k0+15, k0+24..k0+31}
DEVINL v16bf a_frag(const __bf16* __restrict__ A, int lda, int row, int k0, int lane) {
  const int kb = k0 + ((lane >> 4) << 3);
  const __bf16* p = A + (size_t)row * lda + kb;
  v8bf lo = *(const v8bf*)(p);
  v8bf hi = *(const v8bf*)(p + 16);
  return __builtin_shufflevector(lo, hi, 0,1,2,3,4,5,6,7,8,9,10,11,12,13,14,15);
}
// B fragment (32x16 bf16, K x N): lane<16 -> N=lane, K {k0..k0+7, k0+16..k0+23};
// lane>=16 -> N=lane-16, K {k0+8..k0+15, k0+24..k0+31}
DEVINL v16bf b_frag(const __bf16* __restrict__ Bm, int ldb, int k0, int n0, int lane) {
  const int kb = k0 + ((lane >> 4) << 3);
  const __bf16* p = Bm + (size_t)kb * ldb + n0 + (lane & 15);
  v16bf r;
#pragma unroll
  for (int s = 0; s < 8; ++s) {
    r[s]     = p[(size_t)s * ldb];
    r[s + 8] = p[(size_t)(s + 16) * ldb];
  }
  return r;
}
DEVINL v8f wmma_bf16(v16bf a, v16bf b, v8f c) {
  return __builtin_amdgcn_wmma_f32_16x16x32_bf16(false, a, false, b, (short)0, c, false, false);
}

// ---- CDNA5 async copy global->LDS (ASYNCcnt path), with safe fallback ----
#if defined(__AMDGCN__) && __has_builtin(__builtin_amdgcn_global_load_async_to_lds_b128) && \
    __has_builtin(__builtin_amdgcn_s_wait_asynccnt)
#define USE_ASYNC_LDS 1
#else
#define USE_ASYNC_LDS 0
#endif

DEVINL void stage16(__bf16* dst_lds, const __bf16* src_glb) {
#if USE_ASYNC_LDS
  __builtin_amdgcn_global_load_async_to_lds_b128(
      (__attribute__((address_space(1))) v4i*)src_glb,
      (__attribute__((address_space(3))) v4i*)dst_lds, 0, 0);
#else
  *(v8bf*)dst_lds = *(const v8bf*)src_glb;
#endif
}
DEVINL void stage_commit() {
#if USE_ASYNC_LDS
  __builtin_amdgcn_s_wait_asynccnt(0);
#endif
  __syncthreads();
}

// ---------------- prep kernels ----------------
__global__ __launch_bounds__(256) void k_transpose_bf16(const float* __restrict__ src,
                                                        __bf16* __restrict__ dst,
                                                        int R, int C) {
  size_t i = (size_t)blockIdx.x * blockDim.x + threadIdx.x;
  if (i >= (size_t)R * C) return;
  int r = (int)(i / C), c = (int)(i % C);
  dst[(size_t)c * R + r] = (__bf16)src[i];
}

__global__ __launch_bounds__(256) void k_wdec_pack(const float* __restrict__ W_dec,
                                                   const float* __restrict__ W_halt,
                                                   __bf16* __restrict__ WdecT) {
  size_t i = (size_t)blockIdx.x * blockDim.x + threadIdx.x;
  if (i >= (size_t)Hc * 1024) return;
  int h = (int)(i >> 10), c = (int)(i & 1023);
  float v = 0.0f;
  if (c < NCc)        v = W_dec[(size_t)c * Hc + h];
  else if (c == NCc)  v = W_halt[h];
  WdecT[i] = (__bf16)v;
}

__global__ __launch_bounds__(256) void k_x_to_bf16(const float* __restrict__ x,
                                                   __bf16* __restrict__ xbf) {
  size_t i = (size_t)blockIdx.x * blockDim.x + threadIdx.x;
  if (i >= (size_t)Tc * Bc * Dc) return;
  int t = (int)(i / ((size_t)Bc * Dc));
  int rem = (int)(i % ((size_t)Bc * Dc));
  int b = rem / Dc, d = rem % Dc;
  xbf[i] = (__bf16)x[((size_t)b * Dc + d) * Tc + t];
}

__global__ __launch_bounds__(256) void k_bias(const float* __restrict__ bih,
                                              const float* __restrict__ bhh,
                                              const float* __restrict__ b_dec,
                                              const float* __restrict__ b_halt,
                                              float* __restrict__ bsum,
                                              float* __restrict__ bdec) {
  int j = blockIdx.x * blockDim.x + threadIdx.x;
  if (j < G4) bsum[j] = bih[j] + bhh[j];
  if (j < 1024) {
    float v = 0.0f;
    if (j < NCc) v = b_dec[j];
    else if (j == NCc) v = b_halt[0];
    bdec[j] = v;
  }
}

__global__ __launch_bounds__(256) void k_init(__bf16* __restrict__ h0,
                                              float* __restrict__ c_cur,
                                              float* __restrict__ P) {
  int i = blockIdx.x * blockDim.x + threadIdx.x;
  if (i < Bc * Hc) { h0[i] = (__bf16)0.0f; c_cur[i] = 0.0f; }
  if (i < Bc) P[i] = 0.0f;
}

// ---------------- x-projection GEMM: Xproj[B,4H] = xbf_t[B,D] @ WihT[D,4H] + bsum ----
__global__ __launch_bounds__(256) void k_xproj(const __bf16* __restrict__ Axt,
                                               const __bf16* __restrict__ WihT,
                                               const float* __restrict__ bsum,
                                               float* __restrict__ Xproj) {
  const int lane = threadIdx.x & 31;
  const int wv   = threadIdx.x >> 5;
  const int m0   = blockIdx.x * 128 + wv * 16;
  const int n0   = blockIdx.y * 64;
  const int arow = m0 + (lane & 15);
  v8f a0 = {}, a1 = {}, a2 = {}, a3 = {};
  for (int k0 = 0; k0 < Dc; k0 += 32) {
    v16bf a = a_frag(Axt, Dc, arow, k0, lane);
    a0 = wmma_bf16(a, b_frag(WihT, G4, k0, n0,      lane), a0);
    a1 = wmma_bf16(a, b_frag(WihT, G4, k0, n0 + 16, lane), a1);
    a2 = wmma_bf16(a, b_frag(WihT, G4, k0, n0 + 32, lane), a2);
    a3 = wmma_bf16(a, b_frag(WihT, G4, k0, n0 + 48, lane), a3);
  }
  const int nn = lane & 15, mo = (lane >> 4) << 3;
#pragma unroll
  for (int r = 0; r < 8; ++r) {
    const size_t row = (size_t)(m0 + r + mo) * G4;
    Xproj[row + n0      + nn] = a0[r] + bsum[n0      + nn];
    Xproj[row + n0 + 16 + nn] = a1[r] + bsum[n0 + 16 + nn];
    Xproj[row + n0 + 32 + nn] = a2[r] + bsum[n0 + 32 + nn];
    Xproj[row + n0 + 48 + nn] = a3[r] + bsum[n0 + 48 + nn];
  }
}

// ---------------- fused recurrent GEMM + LSTM cell ----------------
// Block: 8 waves x (16 batch rows) sharing one 16-col hidden tile.
// The Whh tile (32 K x 64 N = 4 gate groups x 16 cols) is staged in LDS,
// double-buffered, filled via the CDNA5 async global->LDS path when available.
constexpr int SB_STRIDE = 72;  // bf16 elems per row (64 + pad, keeps 16B align)

__global__ __launch_bounds__(256) void k_gate_cell(const __bf16* __restrict__ h_in,
                                                   const __bf16* __restrict__ WhhT,
                                                   const float* __restrict__ Xproj,
                                                   float* __restrict__ c_cur,
                                                   float* __restrict__ s_hist_k,
                                                   float* __restrict__ c_hist_k,
                                                   __bf16* __restrict__ h_out) {
  __shared__ __bf16 sB[2][32 * SB_STRIDE];
  const int tid  = threadIdx.x;
  const int lane = tid & 31;
  const int wv   = tid >> 5;
  const int m0   = blockIdx.x * 128 + wv * 16;   // batch tile
  const int h0   = blockIdx.y * 16;              // hidden-col tile
  const int arow = m0 + (lane & 15);

  // staging geometry: 256 threads x 16B = 32(K) x 64(N) bf16 tile
  const int sk    = tid >> 3;        // K row 0..31
  const int chunk = tid & 7;         // 8B-elem chunk 0..7
  const int ggrp  = chunk >> 1;      // gate group 0..3
  const size_t gsrc = (size_t)sk * G4 + (size_t)ggrp * Hc + h0 + ((chunk & 1) << 3);
  __bf16* sdst[2] = { &sB[0][sk * SB_STRIDE + chunk * 8],
                      &sB[1][sk * SB_STRIDE + chunk * 8] };

  stage16(sdst[0], WhhT + gsrc);     // prologue: slot 0 <- k0 = 0

  v8f ai = {}, af = {}, ag = {}, ao = {};
  for (int k0 = 0; k0 < Hc; k0 += 32) {
    const int s = (k0 >> 5) & 1;
    stage_commit();                  // slot s ready for everyone
    if (k0 + 32 < Hc) stage16(sdst[s ^ 1], WhhT + gsrc + (size_t)(k0 + 32) * G4);

    v16bf a  = a_frag(h_in, Hc, arow, k0, lane);
    const __bf16* sb = &sB[s][((lane >> 4) << 3) * SB_STRIDE + (lane & 15)];
    v16bf b0, b1, b2, b3;
#pragma unroll
    for (int e = 0; e < 16; ++e) {
      const int kr = (e < 8) ? e : e + 8;
      b0[e] = sb[kr * SB_STRIDE];
      b1[e] = sb[kr * SB_STRIDE + 16];
      b2[e] = sb[kr * SB_STRIDE + 32];
      b3[e] = sb[kr * SB_STRIDE + 48];
    }
    ai = wmma_bf16(a, b0, ai);
    af = wmma_bf16(a, b1, af);
    ag = wmma_bf16(a, b2, ag);
    ao = wmma_bf16(a, b3, ao);
  }

  const int hcol = h0 + (lane & 15);
  const int mo   = (lane >> 4) << 3;
#pragma unroll
  for (int r = 0; r < 8; ++r) {
    const int b = m0 + r + mo;
    const size_t gi = (size_t)b * G4;
    const float pi = ai[r] + Xproj[gi +          hcol];
    const float pf = af[r] + Xproj[gi +     Hc + hcol];
    const float pg = ag[r] + Xproj[gi + 2 * Hc + hcol];
    const float po = ao[r] + Xproj[gi + 3 * Hc + hcol];
    const size_t hi = (size_t)b * Hc + hcol;
    const float cp = c_cur[hi];
    const float c2 = sigf(pf) * cp + sigf(pi) * tanhf(pg);
    const float h2 = sigf(po) * tanhf(c2);
    c_cur[hi]    = c2;
    c_hist_k[hi] = c2;
    s_hist_k[hi] = h2;
    h_out[hi]    = (__bf16)h2;
  }
}

// ---------------- decode GEMM: y[B,1024pad] = h2[B,H] @ WdecT[H,1024] + bdec ----
__global__ __launch_bounds__(256) void k_decode(const __bf16* __restrict__ h2,
                                                const __bf16* __restrict__ WdecT,
                                                const float* __restrict__ bdec,
                                                float* __restrict__ yk,
                                                float* __restrict__ pk) {
  const int lane = threadIdx.x & 31;
  const int wv   = threadIdx.x >> 5;
  const int m0   = blockIdx.x * 128 + wv * 16;
  const int n0   = blockIdx.y * 64;
  const int arow = m0 + (lane & 15);
  v8f a0 = {}, a1 = {}, a2 = {}, a3 = {};
  for (int k0 = 0; k0 < Hc; k0 += 32) {
    v16bf a = a_frag(h2, Hc, arow, k0, lane);
    a0 = wmma_bf16(a, b_frag(WdecT, 1024, k0, n0,      lane), a0);
    a1 = wmma_bf16(a, b_frag(WdecT, 1024, k0, n0 + 16, lane), a1);
    a2 = wmma_bf16(a, b_frag(WdecT, 1024, k0, n0 + 32, lane), a2);
    a3 = wmma_bf16(a, b_frag(WdecT, 1024, k0, n0 + 48, lane), a3);
  }
  const int nn = lane & 15, mo = (lane >> 4) << 3;
#pragma unroll
  for (int r = 0; r < 8; ++r) {
    const int b = m0 + r + mo;
#pragma unroll
    for (int s = 0; s < 4; ++s) {
      const int n = n0 + 16 * s + nn;
      float v = (s == 0 ? a0[r] : s == 1 ? a1[r] : s == 2 ? a2[r] : a3[r]) + bdec[n];
      if (n < NCc)       yk[(size_t)b * 1024 + n] = v;
      else if (n == NCc) pk[b] = sigf(v);
    }
  }
}

// ---------------- halting combine: one block per batch row ----------------
__global__ __launch_bounds__(256) void k_combine(const float* __restrict__ s_hist,
                                                 const float* __restrict__ c_hist,
                                                 const float* __restrict__ y_hist,
                                                 const float* __restrict__ p_hist,
                                                 int t,
                                                 float* __restrict__ Y,
                                                 float* __restrict__ P,
                                                 float* __restrict__ Nout,
                                                 float* __restrict__ Hs,
                                                 __bf16* __restrict__ h0buf,
                                                 float* __restrict__ c_cur) {
  const int b = blockIdx.x;
  float pm[Kc], ph[Kc];
  float cum = 0.0f;
#pragma unroll
  for (int kk = 0; kk < Kc; ++kk) {
    cum += p_hist[kk * Bc + b];
    pm[kk] = fminf(1.0f, cum);
  }
  pm[Kc - 1] = 1.0f;
  ph[0] = pm[0];
#pragma unroll
  for (int kk = 1; kk < Kc; ++kk) ph[kk] = pm[kk] - pm[kk - 1];
  int nt = Kc - 1;
#pragma unroll
  for (int kk = Kc - 1; kk >= 0; --kk) if (pm[kk] >= 1.0f) nt = kk;
  const float rt = (nt == 0) ? 0.0f : (1.0f - pm[nt - 1]);
  if (threadIdx.x == 0) {
    Nout[(size_t)b * Tc + t] = (float)nt;
    P[b] += (float)nt + rt;
  }
  for (int hh = threadIdx.x; hh < Hc; hh += 256) {
    const size_t base = (size_t)b * Hc + hh;
    float st = 0.0f, ct = 0.0f;
#pragma unroll
    for (int kk = 0; kk < Kc; ++kk) {
      const float sv = s_hist[(size_t)kk * Bc * Hc + base];
      st += ph[kk] * sv;
      ct += ph[kk] * c_hist[(size_t)kk * Bc * Hc + base];
      Hs[(base * Kc + kk) * Tc + t] = sv;   // Hs[b][h][k][t]
    }
    h0buf[base] = (__bf16)st;
    c_cur[base] = ct;
  }
  for (int cc = threadIdx.x; cc < NCc; cc += 256) {
    float yt = 0.0f;
#pragma unroll
    for (int kk = 0; kk < Kc; ++kk)
      yt += ph[kk] * y_hist[(size_t)kk * Bc * 1024 + (size_t)b * 1024 + cc];
    Y[((size_t)b * NCc + cc) * Tc + t] = yt;
  }
}

// ---------------- host launch ----------------
extern "C" void kernel_launch(void* const* d_in, const int* in_sizes, int n_in,
                              void* d_out, int out_size, void* d_ws, size_t ws_size,
                              hipStream_t stream) {
  const float* x      = (const float*)d_in[0];
  const float* Wih    = (const float*)d_in[1];
  const float* Whh    = (const float*)d_in[2];
  const float* bih    = (const float*)d_in[3];
  const float* bhh    = (const float*)d_in[4];
  const float* W_dec  = (const float*)d_in[5];
  const float* b_dec  = (const float*)d_in[6];
  const float* W_halt = (const float*)d_in[7];
  const float* b_halt = (const float*)d_in[8];

  char* ws = (char*)d_ws;
  __bf16* WihT  = (__bf16*)(ws + OFF_WIHT);
  __bf16* WhhT  = (__bf16*)(ws + OFF_WHHT);
  __bf16* WdecT = (__bf16*)(ws + OFF_WDECT);
  __bf16* xbf   = (__bf16*)(ws + OFF_XBF);
  float*  bsum  = (float*)(ws + OFF_BSUM);
  float*  bdec  = (float*)(ws + OFF_BDEC);
  float*  Xproj = (float*)(ws + OFF_XPROJ);
  __bf16* hbuf0 = (__bf16*)(ws + OFF_H0);
  __bf16* hbuf1 = (__bf16*)(ws + OFF_H1);
  float*  c_cur = (float*)(ws + OFF_CCUR);
  float*  s_hist = (float*)(ws + OFF_SH);
  float*  c_hist = (float*)(ws + OFF_CH);
  float*  y_hist = (float*)(ws + OFF_YH);
  float*  p_hist = (float*)(ws + OFF_PH);

  float* out = (float*)d_out;
  float* Y    = out + OUT_Y;
  float* P    = out + OUT_P;
  float* Nout = out + OUT_N;
  float* Hs   = out + OUT_HS;

  // ---- prep (cheap, rerun every call for determinism) ----
  {
    size_t n = (size_t)G4 * Dc;
    k_transpose_bf16<<<(unsigned)((n + 255) / 256), 256, 0, stream>>>(Wih, WihT, G4, Dc);
    n = (size_t)G4 * Hc;
    k_transpose_bf16<<<(unsigned)((n + 255) / 256), 256, 0, stream>>>(Whh, WhhT, G4, Hc);
    n = (size_t)Hc * 1024;
    k_wdec_pack<<<(unsigned)((n + 255) / 256), 256, 0, stream>>>(W_dec, W_halt, WdecT);
    n = (size_t)Tc * Bc * Dc;
    k_x_to_bf16<<<(unsigned)((n + 255) / 256), 256, 0, stream>>>(x, xbf);
    k_bias<<<(G4 + 255) / 256, 256, 0, stream>>>(bih, bhh, b_dec, b_halt, bsum, bdec);
    k_init<<<(Bc * Hc + 255) / 256, 256, 0, stream>>>(hbuf0, c_cur, P);
  }

  // ---- sequential ACT-LSTM ----
  for (int t = 0; t < Tc; ++t) {
    k_xproj<<<dim3(Bc / 128, G4 / 64), 256, 0, stream>>>(
        xbf + (size_t)t * Bc * Dc, WihT, bsum, Xproj);
    for (int k = 0; k < Kc; ++k) {
      const __bf16* hin  = (k & 1) ? hbuf1 : hbuf0;
      __bf16*       hout = (k & 1) ? hbuf0 : hbuf1;
      k_gate_cell<<<dim3(Bc / 128, Hc / 16), 256, 0, stream>>>(
          hin, WhhT, Xproj, c_cur,
          s_hist + (size_t)k * Bc * Hc, c_hist + (size_t)k * Bc * Hc, hout);
      k_decode<<<dim3(Bc / 128, 1024 / 64), 256, 0, stream>>>(
          hout, WdecT, bdec, y_hist + (size_t)k * Bc * 1024, p_hist + (size_t)k * Bc);
    }
    k_combine<<<Bc, 256, 0, stream>>>(s_hist, c_hist, y_hist, p_hist, t,
                                      Y, P, Nout, Hs, hbuf0, c_cur);
  }
}
